// ProcessRewardModel_58480274702560
// MI455X (gfx1250) — matrix operand order repaired and checked by reading, not compiled, not run on hardware
//
#include <hip/hip_runtime.h>
#include <hip/hip_bf16.h>

// ---------------------------------------------------------------------------
// ProcessRewardModel for MI455X (gfx1250, wave32, WMMA).
// B=8, T=1024, V=50000, H=768, HD=384, S=16.
//
// Pipeline:
//   1. convert LSTM weights f32->bf16 (workspace)
//   2. embedding gather -> X bf16 [8192,768]
//   3. per (layer,dir): pre = X @ W_ih^T via LDS-tiled WMMA GEMM
//      (128x128 block tile, BK=64, double-buffered LDS staged with
//       GLOBAL_LOAD_ASYNC_TO_LDS_B128 — confirmed present on this toolchain)
//   4. per layer: persistent scan kernel (2 blocks = 2 directions),
//      h @ W_hh^T via WMMA from LDS-resident bf16 h, gate nonlinearity
//   5. segment mean-pool + MLP + sigmoid -> step_rewards
//   6. masked min -> total_reward
// ---------------------------------------------------------------------------

typedef __attribute__((ext_vector_type(16))) __bf16 v16bf;
typedef __attribute__((ext_vector_type(8)))  float  v8f;

#define Bsz 8
#define Tsz 1024
#define Hsz 768
#define HD  384
#define G4  1536          // 4*HD
#define MROWS 8192        // B*T
#define Ssz 16

// gfx1250 async LDS copy path (guarded: falls back to plain vector copies)
#if defined(__has_builtin)
# if __has_builtin(__builtin_amdgcn_global_load_async_to_lds_b128)
#  define HAS_ASYNC_LDS 1
# endif
# if __has_builtin(__builtin_amdgcn_s_wait_asynccnt)
#  define HAS_WAIT_ASYNC 1
# endif
#endif
#ifndef HAS_ASYNC_LDS
# define HAS_ASYNC_LDS 0
#endif
#ifndef HAS_WAIT_ASYNC
# define HAS_WAIT_ASYNC 0
#endif

#if HAS_ASYNC_LDS
// exact pointee type per compiler diagnostic: int __attribute__((vector_size(16)))
typedef int async_v4i __attribute__((vector_size(4 * sizeof(int))));
typedef __attribute__((address_space(1))) async_v4i* async_gptr;
typedef __attribute__((address_space(3))) async_v4i* async_lptr;
#endif

__device__ __forceinline__ float sigmoidf(float x) {
    return 1.0f / (1.0f + expf(-x));
}

// ---------------------------------------------------------------------------
// f32 -> bf16 conversion
// ---------------------------------------------------------------------------
__global__ void cvt_f32_bf16(const float* __restrict__ src,
                             __bf16* __restrict__ dst, int n) {
    int i = blockIdx.x * blockDim.x + threadIdx.x;
    if (i < n) dst[i] = (__bf16)src[i];
}

// ---------------------------------------------------------------------------
// embedding gather: xb[row, :] = bf16(emb[ids[row], :])
// ---------------------------------------------------------------------------
__global__ void embed_gather(const int* __restrict__ ids,
                             const float* __restrict__ emb,
                             __bf16* __restrict__ xb) {
    int row = blockIdx.x;                 // 0..8191  (= b*T + t)
    int id  = ids[row];
    const float* e = emb + (size_t)id * Hsz;
    __bf16*      o = xb  + (size_t)row * Hsz;
    for (int d = threadIdx.x; d < Hsz; d += blockDim.x)
        o[d] = (__bf16)e[d];
}

// ---------------------------------------------------------------------------
// LDS-tiled GEMM: Y[M,N] = X[M,K](bf16) @ W[N,K](bf16)^T, f32 accumulate.
// Block tile 128x128, K-chunk 64, double-buffered LDS (async staged).
// 8 waves/block; wave (mg,ng) owns a 64x32 region = 4x2 16x16 subtiles.
// ---------------------------------------------------------------------------
#define BM 128
#define BN 128
#define BK 64

__global__ void __launch_bounds__(256)
gemm_bf16_wmma_tiled(const __bf16* __restrict__ X,   // [M,K]
                     const __bf16* __restrict__ W,   // [N,K] (row n = output col)
                     float* __restrict__ Y,          // [M,N]
                     int M, int N, int K) {
    __shared__ __bf16 As[2][BM * BK];   // 16KB per buffer
    __shared__ __bf16 Bs[2][BN * BK];

    const int tid  = threadIdx.x;
    const int nblk = N / BN;
    const int mb   = blockIdx.x / nblk;
    const int nb   = blockIdx.x - mb * nblk;
    const __bf16* Ag = X + (size_t)mb * BM * K;
    const __bf16* Bg = W + (size_t)nb * BN * K;

    // stage one 128x64 bf16 chunk of A and B into LDS buffer `buf`
    // (16KB each; 256 threads x 4 x 16B per matrix)
    auto issue = [&](int buf, int kc) {
        const int k0 = kc * BK;
#pragma unroll
        for (int i = 0; i < 4; ++i) {
            const int off = (tid + i * 256) * 16;     // byte offset in chunk
            const int row = off >> 7;                 // 128B per row (64 bf16)
            const int ine = (off & 127) >> 1;         // element within row
            const __bf16* ga = Ag + (size_t)row * K + k0 + ine;
            const __bf16* gb = Bg + (size_t)row * K + k0 + ine;
            __bf16* la = &As[buf][row * BK + ine];
            __bf16* lb = &Bs[buf][row * BK + ine];
#if HAS_ASYNC_LDS
            __builtin_amdgcn_global_load_async_to_lds_b128(
                (async_gptr)(void*)ga, (async_lptr)(void*)la, 0, 0);
            __builtin_amdgcn_global_load_async_to_lds_b128(
                (async_gptr)(void*)gb, (async_lptr)(void*)lb, 0, 0);
#else
            *(uint4*)la = *(const uint4*)ga;
            *(uint4*)lb = *(const uint4*)gb;
#endif
        }
    };

    const int wave = tid >> 5;
    const int lane = tid & 31;
    const int hf   = lane >> 4;
    const int lr   = lane & 15;
    const int mg   = wave >> 2;     // 0..1 : M-subtiles mg*4 .. mg*4+3
    const int ng   = wave & 3;      // 0..3 : N-subtiles ng*2, ng*2+1

    v8f acc[4][2] = {};

    issue(0, 0);
    const int nkc = K / BK;                 // 12
    for (int kc = 0; kc < nkc; ++kc) {
#if HAS_WAIT_ASYNC
        __builtin_amdgcn_s_wait_asynccnt(0);
#endif
        __syncthreads();                    // chunk kc resident in LDS
        if (kc + 1 < nkc) issue((kc + 1) & 1, kc + 1);
        const __bf16* Ab = As[kc & 1];
        const __bf16* Bb = Bs[kc & 1];
#pragma unroll
        for (int ks = 0; ks < 2; ++ks) {
            const int k0 = ks * 32;
            v16bf af[4], bfr[2];
#pragma unroll
            for (int mi = 0; mi < 4; ++mi) {
                const int row = (mg * 4 + mi) * 16 + lr;
#pragma unroll
                for (int e = 0; e < 16; ++e)
                    // A 16-bit 16x32 layout: K = hf*16 + e (contiguous, matches B)
                    af[mi][e] = Ab[row * BK + k0 + hf * 16 + e];
            }
#pragma unroll
            for (int ni = 0; ni < 2; ++ni) {
                const int row = (ng * 2 + ni) * 16 + lr;
#pragma unroll
                for (int e = 0; e < 16; ++e)
                    // B 16-bit 32x16 layout: K = hf*16 + e (contiguous)
                    bfr[ni][e] = Bb[row * BK + k0 + hf * 16 + e];
            }
#pragma unroll
            for (int mi = 0; mi < 4; ++mi)
#pragma unroll
                for (int ni = 0; ni < 2; ++ni)
                    acc[mi][ni] = __builtin_amdgcn_wmma_f32_16x16x32_bf16(
                        false, af[mi], false, bfr[ni], (short)0,
                        acc[mi][ni], false, false);
        }
        __syncthreads();                    // done with buffer (kc&1)
    }

    // C/D layout: vgpr r -> row (r + hf*8), col = lr
#pragma unroll
    for (int mi = 0; mi < 4; ++mi) {
        const int grow0 = mb * BM + (mg * 4 + mi) * 16;
#pragma unroll
        for (int ni = 0; ni < 2; ++ni) {
            float* yt = Y + (size_t)grow0 * N + nb * BN + (ng * 2 + ni) * 16 + lr;
#pragma unroll
            for (int r = 0; r < 8; ++r)
                yt[(size_t)(r + hf * 8) * N] = acc[mi][ni][r];
        }
    }
}

// ---------------------------------------------------------------------------
// Persistent bidirectional LSTM scan for one layer.
// grid = 2 (dir 0 = fwd, dir 1 = bwd), block = 256 (8 waves).
// pre: [2][8192][1536] f32 (dir-major), whh_*: [1536][384] bf16.
// enc: [8192][768] f32 out, xb: [8192][768] bf16 out (next layer input).
// ---------------------------------------------------------------------------
__global__ void __launch_bounds__(256)
lstm_scan(const float* __restrict__ pre,
          const __bf16* __restrict__ whh_fwd,
          const __bf16* __restrict__ whh_bwd,
          const float* __restrict__ bih_fwd, const float* __restrict__ bhh_fwd,
          const float* __restrict__ bih_bwd, const float* __restrict__ bhh_bwd,
          float* __restrict__ enc,
          __bf16* __restrict__ xb) {
    const int dir = blockIdx.x;
    const float*  preD = pre + (size_t)dir * MROWS * G4;
    const __bf16* whh  = dir ? whh_bwd : whh_fwd;
    const float*  bih  = dir ? bih_bwd : bih_fwd;
    const float*  bhh  = dir ? bhh_bwd : bhh_fwd;

    __shared__ __bf16 Hb[16 * HD];    // padded-batch hidden state (rows 8..15 = 0)
    __shared__ float  Hp[8 * G4];     // recurrent gate contributions (real rows)

    const int tid = threadIdx.x;
    for (int i = tid; i < 16 * HD; i += 256) Hb[i] = (__bf16)0.0f;

    // each thread owns fixed (b,j) units u = tid + k*256, k = 0..11
    float creg[12];
#pragma unroll
    for (int k = 0; k < 12; ++k) creg[k] = 0.0f;
    __syncthreads();

    const int wave = tid >> 5;
    const int lane = tid & 31;
    const int hf   = lane >> 4;
    const int lr   = lane & 15;

    for (int t = 0; t < Tsz; ++t) {
        const int tt = dir ? (Tsz - 1 - t) : t;

        // ---- recurrent micro-GEMM: Hp[0:8, 0:1536] = Hb @ whh^T ----
        for (int nt = wave; nt < (G4 >> 4); nt += 8) {     // 12 tiles / wave
            v8f acc = {};
            const __bf16* wrow = whh + (size_t)(nt * 16 + lr) * HD;
#pragma unroll 1
            for (int ks = 0; ks < HD / 32; ++ks) {
                const int k0 = ks * 32;
                __builtin_prefetch(wrow + k0 + 64, 0, 1);
                v16bf a, b;
#pragma unroll
                for (int e = 0; e < 16; ++e) {
                    a[e] = Hb[lr * HD + k0 + hf * 16 + e];
                    b[e] = wrow[k0 + hf * 16 + e];
                }
                acc = __builtin_amdgcn_wmma_f32_16x16x32_bf16(
                          false, a, false, b, (short)0, acc, false, false);
            }
            if (hf == 0) {                    // rows 0..7 live in lanes 0-15
#pragma unroll
                for (int r = 0; r < 8; ++r)
                    Hp[r * G4 + nt * 16 + lr] = acc[r];
            }
        }
        __syncthreads();

        // ---- gate nonlinearity + state update (8*384 units) ----
#pragma unroll 1
        for (int k = 0; k < 12; ++k) {
            const int u = tid + k * 256;
            const int b = u / HD;
            const int j = u - b * HD;
            const size_t row = (size_t)b * Tsz + tt;
            const float* pr = preD + row * G4;
            const float gi = Hp[b * G4 + j]          + pr[j]          + bih[j]          + bhh[j];
            const float gf = Hp[b * G4 + HD + j]     + pr[HD + j]     + bih[HD + j]     + bhh[HD + j];
            const float gg = Hp[b * G4 + 2 * HD + j] + pr[2 * HD + j] + bih[2 * HD + j] + bhh[2 * HD + j];
            const float go = Hp[b * G4 + 3 * HD + j] + pr[3 * HD + j] + bih[3 * HD + j] + bhh[3 * HD + j];
            const float c = sigmoidf(gf) * creg[k] + sigmoidf(gi) * tanhf(gg);
            const float h = sigmoidf(go) * tanhf(c);
            creg[k] = c;
            Hb[b * HD + j] = (__bf16)h;
            enc[row * Hsz + dir * HD + j] = h;
            xb [row * Hsz + dir * HD + j] = (__bf16)h;
        }
        __syncthreads();
    }
}

// ---------------------------------------------------------------------------
// segment mean pooling + 2-layer MLP + sigmoid.  grid = B*S, block = 256.
// ---------------------------------------------------------------------------
__global__ void pool_mlp(const float* __restrict__ enc,
                         const int* __restrict__ sb,
                         const float* __restrict__ w1, const float* __restrict__ b1,
                         const float* __restrict__ w2, const float* __restrict__ b2,
                         float* __restrict__ out) {
    const int bs = blockIdx.x;
    const int b  = bs >> 4;
    const int s  = bs & 15;
    __shared__ float mean[Hsz];
    __shared__ float h1[HD];
    __shared__ float partial[256];

    const int end   = sb[b * Ssz + s];
    const int start = (s == 0) ? 0 : sb[b * Ssz + s - 1];
    int cnt = end - start; if (cnt < 1) cnt = 1;
    const float inv = 1.0f / (float)cnt;

    for (int d = threadIdx.x; d < Hsz; d += 256) {
        float acc = 0.0f;
        for (int t = start; t < end; ++t)
            acc += enc[((size_t)b * Tsz + t) * Hsz + d];
        mean[d] = acc * inv;
    }
    __syncthreads();

    for (int j = threadIdx.x; j < HD; j += 256) {
        const float* wr = w1 + (size_t)j * Hsz;
        float acc = b1[j];
        for (int d = 0; d < Hsz; ++d) acc += wr[d] * mean[d];
        h1[j] = fmaxf(acc, 0.0f);
    }
    __syncthreads();

    float p = 0.0f;
    for (int j = threadIdx.x; j < HD; j += 256) p += w2[j] * h1[j];
    partial[threadIdx.x] = p;
    __syncthreads();
    for (int st = 128; st > 0; st >>= 1) {
        if (threadIdx.x < st) partial[threadIdx.x] += partial[threadIdx.x + st];
        __syncthreads();
    }
    if (threadIdx.x == 0)
        out[b * Ssz + s] = sigmoidf(partial[0] + b2[0]);
}

// ---------------------------------------------------------------------------
// total_reward = min over positive step rewards (sigmoid > 0 always).
// ---------------------------------------------------------------------------
__global__ void reward_min(float* __restrict__ out) {
    const int b = threadIdx.x;
    if (b < Bsz) {
        float m = 1.0f;
        for (int s = 0; s < Ssz; ++s) {
            const float r = out[b * Ssz + s];
            const float v = (r > 0.0f) ? r : 1.0f;
            m = fminf(m, v);
        }
        out[Bsz * Ssz + b] = m;
    }
}

// ---------------------------------------------------------------------------
// host side
// ---------------------------------------------------------------------------
extern "C" void kernel_launch(void* const* d_in, const int* in_sizes, int n_in,
                              void* d_out, int out_size, void* d_ws, size_t ws_size,
                              hipStream_t stream) {
    (void)in_sizes; (void)n_in; (void)out_size; (void)ws_size;

    const int*   input_ids = (const int*)d_in[0];
    const int*   sb        = (const int*)d_in[1];
    const float* emb       = (const float*)d_in[2];
    // order per setup_inputs: (l0,f) (l0,b) (l1,f) (l1,b)
    const float* wih[4] = { (const float*)d_in[3],  (const float*)d_in[7],
                            (const float*)d_in[11], (const float*)d_in[15] };
    const float* whh[4] = { (const float*)d_in[4],  (const float*)d_in[8],
                            (const float*)d_in[12], (const float*)d_in[16] };
    const float* bih[4] = { (const float*)d_in[5],  (const float*)d_in[9],
                            (const float*)d_in[13], (const float*)d_in[17] };
    const float* bhh[4] = { (const float*)d_in[6],  (const float*)d_in[10],
                            (const float*)d_in[14], (const float*)d_in[18] };
    const float* w1 = (const float*)d_in[19];
    const float* b1 = (const float*)d_in[20];
    const float* w2 = (const float*)d_in[21];
    const float* b2 = (const float*)d_in[22];
    float* out = (float*)d_out;

    // workspace layout (all offsets multiple of 256B)
    char* ws = (char*)d_ws;
    const size_t XB_BYTES  = (size_t)MROWS * Hsz * 2;        // 12,582,912
    const size_t ENC_BYTES = (size_t)MROWS * Hsz * 4;        // 25,165,824
    const size_t PRE_BYTES = (size_t)2 * MROWS * G4 * 4;     // 100,663,296
    const size_t WIH_BYTES = (size_t)4 * G4 * Hsz * 2;       // 9,437,184
    __bf16* xb   = (__bf16*)(ws);
    float*  enc  = (float*)(ws + XB_BYTES);
    float*  pre  = (float*)(ws + XB_BYTES + ENC_BYTES);
    __bf16* wihb = (__bf16*)(ws + XB_BYTES + ENC_BYTES + PRE_BYTES);
    __bf16* whhb = (__bf16*)(ws + XB_BYTES + ENC_BYTES + PRE_BYTES + WIH_BYTES);

    // 1. weight conversion
    const int nih = G4 * Hsz, nhh = G4 * HD;
    for (int m = 0; m < 4; ++m) {
        hipLaunchKernelGGL(cvt_f32_bf16, dim3((nih + 255) / 256), dim3(256), 0, stream,
                           wih[m], wihb + (size_t)m * nih, nih);
        hipLaunchKernelGGL(cvt_f32_bf16, dim3((nhh + 255) / 256), dim3(256), 0, stream,
                           whh[m], whhb + (size_t)m * nhh, nhh);
    }

    // 2. embedding gather
    hipLaunchKernelGGL(embed_gather, dim3(MROWS), dim3(256), 0, stream,
                       input_ids, emb, xb);

    const int gblocks = (MROWS / BM) * (G4 / BN);   // 64 * 12 = 768

    // 3. layer 0: input projection GEMMs + scan
    hipLaunchKernelGGL(gemm_bf16_wmma_tiled, dim3(gblocks), dim3(256), 0, stream,
                       xb, wihb + 0 * (size_t)nih, pre, MROWS, G4, Hsz);
    hipLaunchKernelGGL(gemm_bf16_wmma_tiled, dim3(gblocks), dim3(256), 0, stream,
                       xb, wihb + 1 * (size_t)nih, pre + (size_t)MROWS * G4,
                       MROWS, G4, Hsz);
    hipLaunchKernelGGL(lstm_scan, dim3(2), dim3(256), 0, stream,
                       pre, whhb + 0 * (size_t)nhh, whhb + 1 * (size_t)nhh,
                       bih[0], bhh[0], bih[1], bhh[1], enc, xb);

    // 4. layer 1: input projection GEMMs + scan (xb now holds layer-0 output)
    hipLaunchKernelGGL(gemm_bf16_wmma_tiled, dim3(gblocks), dim3(256), 0, stream,
                       xb, wihb + 2 * (size_t)nih, pre, MROWS, G4, Hsz);
    hipLaunchKernelGGL(gemm_bf16_wmma_tiled, dim3(gblocks), dim3(256), 0, stream,
                       xb, wihb + 3 * (size_t)nih, pre + (size_t)MROWS * G4,
                       MROWS, G4, Hsz);
    hipLaunchKernelGGL(lstm_scan, dim3(2), dim3(256), 0, stream,
                       pre, whhb + 2 * (size_t)nhh, whhb + 3 * (size_t)nhh,
                       bih[2], bhh[2], bih[3], bhh[3], enc, xb);

    // 5. pooling + MLP -> step_rewards, 6. min -> total_reward
    hipLaunchKernelGGL(pool_mlp, dim3(Bsz * Ssz), dim3(256), 0, stream,
                       enc, sb, w1, b1, w2, b2, out);
    hipLaunchKernelGGL(reward_min, dim3(1), dim3(32), 0, stream, out);
}
